// AlignedEmbeddings_47785806135470
// MI455X (gfx1250) — matrix operand order repaired and checked by reading, not compiled, not run on hardware
//
#include <hip/hip_runtime.h>
#include <math.h>

// ---------------------------------------------------------------------------
// Model dims (fixed by the reference)
// ---------------------------------------------------------------------------
#define N1   128
#define N2   16
#define Edim 256
#define Hdim 512
#define G4H  2048            // 4*H
#define Vocab 32000
#define NCHUNK (Vocab / 64)  // 500 chunks of 64 cols for the lse GEMM

typedef __attribute__((ext_vector_type(16))) _Float16 v16h;
typedef __attribute__((ext_vector_type(8)))  float    v8f;

__device__ __forceinline__ v8f zero8() {
    v8f r;
    #pragma unroll
    for (int i = 0; i < 8; ++i) r[i] = 0.f;
    return r;
}

// ---------------------------------------------------------------------------
// Deterministic counter-based RNG -> N(0,1)  (PCG hash + Box-Muller)
// ---------------------------------------------------------------------------
__device__ __forceinline__ unsigned pcg_hash(unsigned x) {
    x = x * 747796405u + 2891336453u;
    x = ((x >> ((x >> 28) + 4u)) ^ x) * 277803737u;
    return (x >> 22) ^ x;
}
__device__ __forceinline__ float rng_normal(unsigned streamid, unsigned idx) {
    unsigned s = pcg_hash(streamid * 0x9E3779B9u ^ idx);
    unsigned a = pcg_hash(s ^ 0x85EBCA6Bu);
    unsigned b = pcg_hash(s ^ 0xC2B2AE35u);
    float u1 = ((float)(a >> 8) + 0.5f) * (1.0f / 16777216.0f); // (0,1)
    float u2 = ((float)(b >> 8) + 0.5f) * (1.0f / 16777216.0f);
    return sqrtf(-2.0f * __logf(u1)) * __cosf(6.28318530718f * u2);
}

// ---------------------------------------------------------------------------
// Small prep kernels
// ---------------------------------------------------------------------------
__global__ void k_zero_acc(float* acc) {
    if (threadIdx.x < 4) acc[threadIdx.x] = 0.f;
}
__global__ void k_add_vec(const float* __restrict__ a, const float* __restrict__ b,
                          float* __restrict__ o, int n) {
    int i = blockIdx.x * blockDim.x + threadIdx.x;
    if (i < n) o[i] = a[i] + b[i];
}
__global__ void k_cvt_f16(const float* __restrict__ src, _Float16* __restrict__ dst, int n) {
    int i = blockIdx.x * blockDim.x + threadIdx.x;
    if (i < n) dst[i] = (_Float16)src[i];
}
__global__ void k_embed(const int* __restrict__ words, const float* __restrict__ emb,
                        float* __restrict__ x) {
    int i = blockIdx.x * blockDim.x + threadIdx.x;     // N1*E threads
    int t = i / Edim, e = i % Edim;
    x[i] = emb[(size_t)words[t] * Edim + e];
}

// ---------------------------------------------------------------------------
// Fragment builders per ISA 7.12.2 layouts (wave32, 16x16x32 f16).
//  A (16x32): lane<16 -> row=m0+lane, K {k0..k0+7, k0+16..k0+23}
//             lane>=16 -> row=m0+lane-16, K {k0+8..k0+15, k0+24..k0+31}
//  B (32x16): lane<16 -> col=n0+lane, K {k0..k0+7, k0+16..k0+23}
//             lane>=16 -> col=n0+lane-16, K {k0+8..k0+15, k0+24..k0+31}
// ---------------------------------------------------------------------------
__device__ __forceinline__ v16h frag_a_f32(const float* __restrict__ A, int ld,
                                           int m0, int k0, int lane) {
    const int row = m0 + (lane & 15);
    const int kb  = k0 + ((lane >= 16) ? 8 : 0);
    const float* ap = A + (size_t)row * ld + kb;
    v16h a;
    #pragma unroll
    for (int j = 0; j < 8; ++j) { a[j] = (_Float16)ap[j]; a[8 + j] = (_Float16)ap[16 + j]; }
    return a;
}
__device__ __forceinline__ v16h frag_a_f16(const _Float16* __restrict__ A, int ld,
                                           int m0, int k0, int lane) {
    const int row = m0 + (lane & 15);
    const int kb  = k0 + ((lane >= 16) ? 8 : 0);
    const _Float16* ap = A + (size_t)row * ld + kb;
    v16h a;
    #pragma unroll
    for (int j = 0; j < 8; ++j) { a[j] = ap[j]; a[8 + j] = ap[16 + j]; }
    return a;
}
__device__ __forceinline__ v16h frag_b_f32(const float* __restrict__ W, int ld,
                                           int col0, int k0, int lane) {
    const int col = col0 + (lane & 15);
    const int kb  = k0 + ((lane >= 16) ? 8 : 0);
    const float* wp = W + (size_t)col * ld + kb;
    v16h b;
    #pragma unroll
    for (int j = 0; j < 8; ++j) { b[j] = (_Float16)wp[j]; b[8 + j] = (_Float16)wp[16 + j]; }
    return b;
}
__device__ __forceinline__ v16h frag_b_f16(const _Float16* __restrict__ W, int ld,
                                           int col0, int k0, int lane) {
    const int col = col0 + (lane & 15);
    const int kb  = k0 + ((lane >= 16) ? 8 : 0);
    const _Float16* wp = W + (size_t)col * ld + kb;
    v16h b;
    #pragma unroll
    for (int j = 0; j < 8; ++j) { b[j] = wp[j]; b[8 + j] = wp[16 + j]; }
    return b;
}

// ---------------------------------------------------------------------------
// Generic WMMA GEMM: C[M,N] = A[M,K] @ W[N,K]^T + bias   (A,W f32; cvt on fly)
// Block = 128 threads (4 waves); block tile 64(M) x 64(N); wave tile 16x64.
// ---------------------------------------------------------------------------
__global__ void __launch_bounds__(128)
k_gemm_f32(const float* __restrict__ A, const float* __restrict__ W,
           const float* __restrict__ bias, float* __restrict__ C,
           int M, int N, int K) {
    const int wave = threadIdx.x >> 5;
    const int lane = threadIdx.x & 31;
    const int m0 = blockIdx.y * 64 + wave * 16;
    const int n0 = blockIdx.x * 64;

    v8f acc[4];
    #pragma unroll
    for (int t = 0; t < 4; ++t) acc[t] = zero8();

    for (int k0 = 0; k0 < K; k0 += 32) {
        const v16h a = frag_a_f32(A, K, m0, k0, lane);
        #pragma unroll
        for (int t = 0; t < 4; ++t) {
            const v16h b = frag_b_f32(W, K, n0 + 16 * t, k0, lane);
            acc[t] = __builtin_amdgcn_wmma_f32_16x16x32_f16(false, a, false, b,
                                                            (short)0, acc[t], false, false);
        }
    }
    #pragma unroll
    for (int t = 0; t < 4; ++t) {
        const int col = n0 + 16 * t + (lane & 15);
        const float bv = bias ? bias[col] : 0.f;
        #pragma unroll
        for (int j = 0; j < 8; ++j) {
            const int row = m0 + ((lane < 16) ? j : 8 + j);
            C[(size_t)row * N + col] = acc[t][j] + bv;
        }
    }
}

// ---------------------------------------------------------------------------
// Bidirectional LSTM scan. One 1024-thread block (32 waves) per direction.
// Recurrent matvec h[512] @ Whh^T[512,2048] on the matrix pipe (row 0 of A).
// ---------------------------------------------------------------------------
__global__ void __launch_bounds__(1024)
k_lstm(const float* __restrict__ xgF, const float* __restrict__ xgB,
       const _Float16* __restrict__ WhF, const _Float16* __restrict__ WhB,
       float* __restrict__ hOut) {
    const int dir = blockIdx.x;
    const float*    xg  = dir ? xgB : xgF;
    const _Float16* Whh = dir ? WhB : WhF;

    __shared__ _Float16 hS[Hdim];
    __shared__ float    gS[G4H];

    const int tid  = threadIdx.x;
    const int wave = tid >> 5;
    const int lane = tid & 31;
    const int n0   = wave * 64;       // this wave's 64 gate outputs

    if (tid < Hdim) hS[tid] = (_Float16)0.f;
    float c = 0.f;
    __syncthreads();

    for (int step = 0; step < N1; ++step) {
        const int t = dir ? (N1 - 1 - step) : step;

        v8f acc[4];
        #pragma unroll
        for (int q = 0; q < 4; ++q) acc[q] = zero8();

        for (int k0 = 0; k0 < Hdim; k0 += 32) {
            v16h a;
            #pragma unroll
            for (int j = 0; j < 16; ++j) a[j] = (_Float16)0.f;
            if (lane == 0 || lane == 16) {
                const int kb = k0 + (lane ? 8 : 0);
                #pragma unroll
                for (int j = 0; j < 8; ++j) { a[j] = hS[kb + j]; a[8 + j] = hS[kb + 16 + j]; }
            }
            #pragma unroll
            for (int q = 0; q < 4; ++q) {
                const v16h b = frag_b_f16(Whh, Hdim, n0 + 16 * q, k0, lane);
                acc[q] = __builtin_amdgcn_wmma_f32_16x16x32_f16(false, a, false, b,
                                                                (short)0, acc[q], false, false);
            }
        }
        if (lane < 16) {                         // row M=0 of D = acc[q][0], lanes 0..15
            #pragma unroll
            for (int q = 0; q < 4; ++q) {
                const int col = n0 + 16 * q + lane;
                gS[col] = acc[q][0] + xg[(size_t)t * G4H + col];
            }
        }
        __syncthreads();
        if (tid < Hdim) {                        // torch gate order: i,f,g,o
            const float gi = gS[tid], gf = gS[Hdim + tid];
            const float gg = gS[2 * Hdim + tid], go = gS[3 * Hdim + tid];
            const float si = 1.f / (1.f + __expf(-gi));
            const float sf = 1.f / (1.f + __expf(-gf));
            const float so = 1.f / (1.f + __expf(-go));
            c = sf * c + si * tanhf(gg);
            const float hv = so * tanhf(c);
            hS[tid] = (_Float16)hv;
            hOut[(size_t)t * (2 * Hdim) + dir * Hdim + tid] = hv;
        }
        __syncthreads();
    }
}

// ---------------------------------------------------------------------------
// Softplus + KL partial + sampling (z1 f16, z2 f16)
// ---------------------------------------------------------------------------
__global__ void __launch_bounds__(256)
k_sample_kl(const float* __restrict__ u, const float* __restrict__ spre,
            _Float16* __restrict__ z1, _Float16* __restrict__ z2,
            float* __restrict__ acc) {
    const int i = blockIdx.x * blockDim.x + threadIdx.x;   // N1*H = 65536
    const float uu = u[i];
    const float sp = spre[i];
    const float s  = (sp > 20.f) ? sp : log1pf(__expf(sp));
    const float kl = 0.5f * (s * s + uu * uu - 1.f - 2.f * __logf(s));
    z1[i] = (_Float16)(uu + s * rng_normal(1u, (unsigned)i));
    #pragma unroll 4
    for (int j = 0; j < N2; ++j)
        z2[(size_t)j * (N1 * Hdim) + i] =
            (_Float16)(uu + s * rng_normal(2u, (unsigned)(j * (N1 * Hdim) + i)));

    __shared__ float red[256];
    red[threadIdx.x] = kl;
    __syncthreads();
    for (int sft = 128; sft > 0; sft >>= 1) {
        if (threadIdx.x < sft) red[threadIdx.x] += red[threadIdx.x + sft];
        __syncthreads();
    }
    if (threadIdx.x == 0) atomicAdd(&acc[0], red[0]);
}

// ---------------------------------------------------------------------------
// Target logit per row: tlog[r] = z[r] . W[word] + b[word]  (1 wave / row)
// ---------------------------------------------------------------------------
__global__ void __launch_bounds__(32)
k_tlogit(const _Float16* __restrict__ Z, const float* __restrict__ W,
         const float* __restrict__ b, const int* __restrict__ words,
         int rowsPerWord, float* __restrict__ tlog) {
    const int row  = blockIdx.x;
    const int lane = threadIdx.x;
    const int w    = words[row / rowsPerWord];
    const _Float16* z = Z + (size_t)row * Hdim;
    const float*    wr = W + (size_t)w * Hdim;
    float d = 0.f;
    for (int k = lane; k < Hdim; k += 32) d += (float)z[k] * wr[k];
    #pragma unroll
    for (int m = 16; m > 0; m >>= 1) d += __shfl_xor(d, m, 32);
    if (lane == 0) tlog[row] = d + b[w];
}

// ---------------------------------------------------------------------------
// Dominant kernel: logits = Z[M,512] @ W[V,512]^T + bias with fused online
// per-64-col (max, sumexp) epilogue. Two variants: W in f16 (fast path,
// pure load_b128 + wmma inner loop) or f32 (cvt on the fly fallback).
// Block = 128 threads (4 waves); block tile 64x64; wave tile 16x64.
// ---------------------------------------------------------------------------
__device__ __forceinline__ void lse_epilogue(v8f* acc, const float* __restrict__ bias,
                                             float2* __restrict__ part,
                                             int m0, int n0, int lane, int nchunks,
                                             int chunk) {
    float bv[4];
    #pragma unroll
    for (int t = 0; t < 4; ++t) bv[t] = bias[n0 + 16 * t + (lane & 15)];
    #pragma unroll
    for (int j = 0; j < 8; ++j) {
        float m = -1e30f;
        #pragma unroll
        for (int t = 0; t < 4; ++t) m = fmaxf(m, acc[t][j] + bv[t]);
        #pragma unroll
        for (int msk = 1; msk < 16; msk <<= 1) m = fmaxf(m, __shfl_xor(m, msk, 32));
        float ss = 0.f;
        #pragma unroll
        for (int t = 0; t < 4; ++t) ss += __expf(acc[t][j] + bv[t] - m);
        #pragma unroll
        for (int msk = 1; msk < 16; msk <<= 1) ss += __shfl_xor(ss, msk, 32);
        if ((lane & 15) == 0) {
            const int row = m0 + ((lane >> 4) ? (8 + j) : j);
            float2 p; p.x = m; p.y = ss;
            part[(size_t)row * nchunks + chunk] = p;
        }
    }
}

__global__ void __launch_bounds__(128)
k_gemm_lse_h(const _Float16* __restrict__ Z, const _Float16* __restrict__ Wh,
             const float* __restrict__ bias, float2* __restrict__ part) {
    const int wave = threadIdx.x >> 5;
    const int lane = threadIdx.x & 31;
    const int m0 = blockIdx.y * 64 + wave * 16;
    const int n0 = blockIdx.x * 64;

    v8f acc[4];
    #pragma unroll
    for (int t = 0; t < 4; ++t) acc[t] = zero8();

    for (int k0 = 0; k0 < Hdim; k0 += 32) {
        const v16h a = frag_a_f16(Z, Hdim, m0, k0, lane);
        #pragma unroll
        for (int t = 0; t < 4; ++t) {
            const v16h b = frag_b_f16(Wh, Hdim, n0 + 16 * t, k0, lane);
            acc[t] = __builtin_amdgcn_wmma_f32_16x16x32_f16(false, a, false, b,
                                                            (short)0, acc[t], false, false);
        }
    }
    lse_epilogue(acc, bias, part, m0, n0, lane, gridDim.x, blockIdx.x);
}

__global__ void __launch_bounds__(128)
k_gemm_lse_f(const _Float16* __restrict__ Z, const float* __restrict__ W,
             const float* __restrict__ bias, float2* __restrict__ part) {
    const int wave = threadIdx.x >> 5;
    const int lane = threadIdx.x & 31;
    const int m0 = blockIdx.y * 64 + wave * 16;
    const int n0 = blockIdx.x * 64;

    v8f acc[4];
    #pragma unroll
    for (int t = 0; t < 4; ++t) acc[t] = zero8();

    for (int k0 = 0; k0 < Hdim; k0 += 32) {
        const v16h a = frag_a_f16(Z, Hdim, m0, k0, lane);
        #pragma unroll
        for (int t = 0; t < 4; ++t) {
            const v16h b = frag_b_f32(W, Hdim, n0 + 16 * t, k0, lane);
            acc[t] = __builtin_amdgcn_wmma_f32_16x16x32_f16(false, a, false, b,
                                                            (short)0, acc[t], false, false);
        }
    }
    lse_epilogue(acc, bias, part, m0, n0, lane, gridDim.x, blockIdx.x);
}

// ---------------------------------------------------------------------------
// Per-row lse combine + gather term accumulation.
// ---------------------------------------------------------------------------
__global__ void __launch_bounds__(256)
k_lse_reduce(const float2* __restrict__ part, const float* __restrict__ tlog,
             int nchunks, float scale, float* __restrict__ accTerm) {
    const int row = blockIdx.x;
    const int tid = threadIdx.x;
    const size_t base = (size_t)row * nchunks;
    __shared__ float sm[256];

    float m = -1e30f;
    for (int c = tid; c < nchunks; c += 256) m = fmaxf(m, part[base + c].x);
    sm[tid] = m; __syncthreads();
    for (int s = 128; s > 0; s >>= 1) {
        if (tid < s) sm[tid] = fmaxf(sm[tid], sm[tid + s]);
        __syncthreads();
    }
    const float M = sm[0];
    __syncthreads();

    float ss = 0.f;
    for (int c = tid; c < nchunks; c += 256) {
        float2 p = part[base + c];
        ss += p.y * __expf(p.x - M);
    }
    sm[tid] = ss; __syncthreads();
    for (int s = 128; s > 0; s >>= 1) {
        if (tid < s) sm[tid] += sm[tid + s];
        __syncthreads();
    }
    if (tid == 0)
        atomicAdd(accTerm, (tlog[row] - (M + __logf(sm[0]))) * scale);
}

__global__ void k_final(const float* __restrict__ acc, float* __restrict__ out) {
    out[0] = -acc[0] + acc[1] + acc[2];
}

// ---------------------------------------------------------------------------
// Host side
// ---------------------------------------------------------------------------
static inline char* ws_take(char*& p, size_t bytes) {
    char* r = p;
    p += (bytes + 255) & ~(size_t)255;
    return r;
}

extern "C" void kernel_launch(void* const* d_in, const int* in_sizes, int n_in,
                              void* d_out, int out_size, void* d_ws, size_t ws_size,
                              hipStream_t stream) {
    const int*   words1 = (const int*)d_in[0];
    const int*   words2 = (const int*)d_in[1];
    const float* emb    = (const float*)d_in[2];
    const float* WihF   = (const float*)d_in[3];
    const float* WhhF   = (const float*)d_in[4];
    const float* bihF   = (const float*)d_in[5];
    const float* bhhF   = (const float*)d_in[6];
    const float* WihB   = (const float*)d_in[7];
    const float* WhhB   = (const float*)d_in[8];
    const float* bihB   = (const float*)d_in[9];
    const float* bhhB   = (const float*)d_in[10];
    const float* UW     = (const float*)d_in[11];
    const float* Ub     = (const float*)d_in[12];
    const float* SW     = (const float*)d_in[13];
    const float* Sb     = (const float*)d_in[14];
    const float* fW     = (const float*)d_in[15];
    const float* fb     = (const float*)d_in[16];
    const float* gW     = (const float*)d_in[17];
    const float* gb     = (const float*)d_in[18];
    float* out = (float*)d_out;

    // ---- workspace carving ----
    char* p = (char*)d_ws;
    float*     x     = (float*)    ws_take(p, (size_t)N1 * Edim * 4);
    float*     bsF   = (float*)    ws_take(p, G4H * 4);
    float*     bsB   = (float*)    ws_take(p, G4H * 4);
    float*     xgF   = (float*)    ws_take(p, (size_t)N1 * G4H * 4);
    float*     xgB   = (float*)    ws_take(p, (size_t)N1 * G4H * 4);
    _Float16*  whF16 = (_Float16*) ws_take(p, (size_t)G4H * Hdim * 2);
    _Float16*  whB16 = (_Float16*) ws_take(p, (size_t)G4H * Hdim * 2);
    float*     h     = (float*)    ws_take(p, (size_t)N1 * 2 * Hdim * 4);
    float*     u     = (float*)    ws_take(p, (size_t)N1 * Hdim * 4);
    float*     spre  = (float*)    ws_take(p, (size_t)N1 * Hdim * 4);
    _Float16*  z1    = (_Float16*) ws_take(p, (size_t)N1 * Hdim * 2);
    _Float16*  z2    = (_Float16*) ws_take(p, (size_t)N2 * N1 * Hdim * 2);
    float*     tlog1 = (float*)    ws_take(p, N1 * 4);
    float*     tlog2 = (float*)    ws_take(p, (size_t)N2 * N1 * 4);
    float2*    part1 = (float2*)   ws_take(p, (size_t)N1 * NCHUNK * 8);
    float2*    part2 = (float2*)   ws_take(p, (size_t)N2 * N1 * NCHUNK * 8);
    float*     acc   = (float*)    ws_take(p, 4 * 4);
    // optional fast-path: f16 copies of the vocab matrices (+64MB)
    char* p_mark = p;
    _Float16*  fW16  = (_Float16*) ws_take(p, (size_t)Vocab * Hdim * 2);
    _Float16*  gW16  = (_Float16*) ws_take(p, (size_t)Vocab * Hdim * 2);
    const bool fast  = ((size_t)(p - (char*)d_ws) <= ws_size);
    (void)p_mark; (void)n_in; (void)in_sizes; (void)out_size;

    // 0) init + weight prep
    k_zero_acc<<<1, 32, 0, stream>>>(acc);
    k_add_vec<<<(G4H + 255) / 256, 256, 0, stream>>>(bihF, bhhF, bsF, G4H);
    k_add_vec<<<(G4H + 255) / 256, 256, 0, stream>>>(bihB, bhhB, bsB, G4H);
    const int nWhh = G4H * Hdim;
    k_cvt_f16<<<(nWhh + 255) / 256, 256, 0, stream>>>(WhhF, whF16, nWhh);
    k_cvt_f16<<<(nWhh + 255) / 256, 256, 0, stream>>>(WhhB, whB16, nWhh);
    if (fast) {
        const int nV = Vocab * Hdim;
        k_cvt_f16<<<(nV + 255) / 256, 256, 0, stream>>>(fW, fW16, nV);
        k_cvt_f16<<<(nV + 255) / 256, 256, 0, stream>>>(gW, gW16, nV);
    }

    // 1) embedding gather
    k_embed<<<(N1 * Edim) / 256, 256, 0, stream>>>(words1, emb, x);

    // 2) input-gate GEMMs: xg = x @ Wih^T + (bih+bhh)
    k_gemm_f32<<<dim3(G4H / 64, N1 / 64), 128, 0, stream>>>(x, WihF, bsF, xgF, N1, G4H, Edim);
    k_gemm_f32<<<dim3(G4H / 64, N1 / 64), 128, 0, stream>>>(x, WihB, bsB, xgB, N1, G4H, Edim);

    // 3) bidirectional recurrent scan (WMMA matvec), writes h [N1, 2H]
    k_lstm<<<2, 1024, 0, stream>>>(xgF, xgB, whF16, whB16, h);

    // 4) u, s_pre GEMMs
    k_gemm_f32<<<dim3(Hdim / 64, N1 / 64), 128, 0, stream>>>(h, UW, Ub, u,    N1, Hdim, 2 * Hdim);
    k_gemm_f32<<<dim3(Hdim / 64, N1 / 64), 128, 0, stream>>>(h, SW, Sb, spre, N1, Hdim, 2 * Hdim);

    // 5) softplus + KL + sampling
    k_sample_kl<<<(N1 * Hdim) / 256, 256, 0, stream>>>(u, spre, z1, z2, acc);

    // 6) target logits
    k_tlogit<<<N1, 32, 0, stream>>>(z1, fW, fb, words1, 1, tlog1);
    k_tlogit<<<N2 * N1, 32, 0, stream>>>(z2, gW, gb, words2, N1, tlog2);

    // 7) dominant vocab GEMMs with fused online (max, sumexp) per chunk
    if (fast) {
        k_gemm_lse_h<<<dim3(NCHUNK, N1 / 64),        128, 0, stream>>>(z1, fW16, fb, part1);
        k_gemm_lse_h<<<dim3(NCHUNK, (N2 * N1) / 64), 128, 0, stream>>>(z2, gW16, gb, part2);
    } else {
        k_gemm_lse_f<<<dim3(NCHUNK, N1 / 64),        128, 0, stream>>>(z1, fW, fb, part1);
        k_gemm_lse_f<<<dim3(NCHUNK, (N2 * N1) / 64), 128, 0, stream>>>(z2, gW, gb, part2);
    }

    // 8) per-row lse combine + gathered-term accumulation
    k_lse_reduce<<<N1,      256, 0, stream>>>(part1, tlog1, NCHUNK, 1.0f,             &acc[1]);
    k_lse_reduce<<<N2 * N1, 256, 0, stream>>>(part2, tlog2, NCHUNK, 1.0f / (float)N2, &acc[2]);

    // 9) final scalar: -kl + term1 + term2
    k_final<<<1, 1, 0, stream>>>(acc, out);
}
